// MoE_65481071394962
// MI455X (gfx1250) — compile-verified, matrix operand order checked by reading, not run on hardware
//
#include <hip/hip_runtime.h>

#define TOK   8192
#define DIM   1024
#define INTER 2048
#define NEXP  8

#define KT 32          // K per tile == WMMA K
#define SA 40          // padded LDS row stride (elements) -> conflict-free b128 reads
#define BM 128
#define BN 64

#if defined(__has_builtin)
# if __has_builtin(__builtin_amdgcn_cvt_pk_bf16_f32)
#  define HAVE_CVT_PK_BF16 1
# endif
# if __has_builtin(__builtin_amdgcn_global_load_async_to_lds_b128)
#  define HAVE_ASYNC_LDS 1
# endif
#endif

typedef __attribute__((ext_vector_type(16))) __bf16 v16bf;
typedef __attribute__((ext_vector_type(2)))  __bf16 v2bf;
typedef __attribute__((ext_vector_type(8)))  float  v8f;

union Frag16 { v16bf v; uint4 q[2]; };

__device__ __forceinline__ unsigned int pk_bf16x2(float lo, float hi) {
#ifdef HAVE_CVT_PK_BF16
  v2bf r = __builtin_amdgcn_cvt_pk_bf16_f32(lo, hi);
  return __builtin_bit_cast(unsigned int, r);
#else
  // single v_perm_b32: take high halves (truncate-to-bf16)
  return __builtin_amdgcn_perm(__builtin_bit_cast(unsigned int, hi),
                               __builtin_bit_cast(unsigned int, lo),
                               0x07060302u);
#endif
}
__device__ __forceinline__ unsigned short f2bf(float f) {
  return (unsigned short)(pk_bf16x2(f, f) & 0xFFFFu);
}

#ifdef HAVE_ASYNC_LDS
typedef int v4i __attribute__((vector_size(16)));
typedef __attribute__((address_space(1))) v4i GV4;      // global int4*
typedef __attribute__((address_space(3))) v4i LV4;      // LDS int4*
__device__ __forceinline__ void async_copy16(const unsigned short* g, unsigned short* l) {
  __builtin_amdgcn_global_load_async_to_lds_b128((GV4*)(void*)g, (LV4*)(void*)l, 0, 0);
}
#define ASYNC_WAIT() asm volatile("s_wait_asynccnt 0" ::: "memory")
#else
#define ASYNC_WAIT()
#endif

// ---------------------------------------------------------------- cast x->bf16
__global__ __launch_bounds__(256)
void moe_cast_bf16(const float* __restrict__ src, unsigned short* __restrict__ dst, int n4) {
  int i = blockIdx.x * 256 + threadIdx.x;
  if (i >= n4) return;
  float4 f = ((const float4*)src)[i];
  uint2 u;
  u.x = pk_bf16x2(f.x, f.y);
  u.y = pk_bf16x2(f.z, f.w);
  ((uint2*)dst)[i] = u;
}

// ---------------------------------------------------------------- gate + top-2
__global__ __launch_bounds__(256)
void moe_gate(const float* __restrict__ x, const float* __restrict__ gw,
              const float* __restrict__ gb, float* __restrict__ combine) {
  __shared__ float gws[NEXP * DIM];
  __shared__ float gbs[NEXP];
  for (int i = threadIdx.x; i < NEXP * DIM; i += 256) gws[i] = gw[i];
  if (threadIdx.x < NEXP) gbs[threadIdx.x] = gb[threadIdx.x];
  __syncthreads();

  const int lane = threadIdx.x & 31;
  const int t = blockIdx.x * 8 + (threadIdx.x >> 5);   // one wave32 per token
  const float* xr = x + (size_t)t * DIM;

  float acc[NEXP];
#pragma unroll
  for (int e = 0; e < NEXP; ++e) acc[e] = 0.f;

  for (int j = 0; j < DIM / 32; ++j) {
    const int d = lane + 32 * j;
    const float xv = xr[d];
#pragma unroll
    for (int e = 0; e < NEXP; ++e) acc[e] += xv * gws[e * DIM + d];
  }
#pragma unroll
  for (int e = 0; e < NEXP; ++e) {
#pragma unroll
    for (int m = 16; m >= 1; m >>= 1) acc[e] += __shfl_xor(acc[e], m, 32);
  }
  if (lane < NEXP) {
    float mx = acc[0];
#pragma unroll
    for (int e = 1; e < NEXP; ++e) mx = fmaxf(mx, acc[e]);
    float ex[NEXP]; float s = 0.f;
#pragma unroll
    for (int e = 0; e < NEXP; ++e) { ex[e] = __expf(acc[e] - mx); s += ex[e]; }
    const float inv = 1.f / s;
    float score[NEXP], biased[NEXP];
#pragma unroll
    for (int e = 0; e < NEXP; ++e) { score[e] = ex[e] * inv; biased[e] = score[e] + gbs[e]; }
    int t1 = 0;
#pragma unroll
    for (int e = 1; e < NEXP; ++e) if (biased[e] > biased[t1]) t1 = e;   // first-wins
    int t2 = (t1 == 0) ? 1 : 0;
#pragma unroll
    for (int e = 0; e < NEXP; ++e)
      if (e != t1 && biased[e] > biased[t2]) t2 = e;
    const int e = lane;
    combine[t * NEXP + e] = (e == t1 || e == t2) ? score[e] : 0.f;
  }
}

// ------------------------------------------------- stage 1: H = silu(XW1^T)*(XW3^T)
__global__ __launch_bounds__(256)
void moe_stage1(const unsigned short* __restrict__ Xb,   // [TOK, DIM] bf16
                const float* __restrict__ W1,            // [INTER, DIM]
                const float* __restrict__ W3,            // [INTER, DIM]
                unsigned short* __restrict__ H) {        // [TOK, INTER] bf16
  __shared__ __align__(16) unsigned short As[2][BM * SA];
  __shared__ __align__(16) unsigned short B1s[2][BN * SA];
  __shared__ __align__(16) unsigned short B3s[2][BN * SA];

  const int tid  = threadIdx.x;
  const int lane = tid & 31;
  const int wv   = tid >> 5;                // 0..7
  const int wm   = wv >> 1;                 // 0..3 : 32-row slab
  const int wn   = wv & 1;                  // 0..1 : 32-col slab
  const int rowBase = blockIdx.x * BM;
  const int colBase = blockIdx.y * BN;
  const int brow = tid >> 2;                // 0..63 B row
  const int bq   = tid & 3;                 // 8-float quarter of k-slice

  uint4 b1R, b3R;
  v8f acc1[2][2], acc3[2][2];
  v8f zero = {0.f, 0.f, 0.f, 0.f, 0.f, 0.f, 0.f, 0.f};
#pragma unroll
  for (int i = 0; i < 2; ++i)
#pragma unroll
    for (int j = 0; j < 2; ++j) { acc1[i][j] = zero; acc3[i][j] = zero; }

  auto stage_A = [&](int kt, int buf) {
    const int kb = kt * KT;
#ifdef HAVE_ASYNC_LDS
#pragma unroll
    for (int k = 0; k < 2; ++k) {           // 128 rows x 4 chunks of 16B
      const int idx = tid + 256 * k;
      const int r = idx >> 2, c = idx & 3;
      async_copy16(Xb + (size_t)(rowBase + r) * DIM + kb + c * 8,
                   &As[buf][r * SA + c * 8]);
    }
#else
    const int r = tid >> 1, hc = (tid & 1) * 16;
    const uint4* ga = (const uint4*)(Xb + (size_t)(rowBase + r) * DIM + kb + hc);
    uint4* da = (uint4*)&As[buf][r * SA + hc];
    da[0] = ga[0]; da[1] = ga[1];
#endif
  };
  auto load_B = [&](int kt) {
    const int kb = kt * KT;
    const float* r1 = W1 + (size_t)(colBase + brow) * DIM + kb + bq * 8;
    const float* r3 = W3 + (size_t)(colBase + brow) * DIM + kb + bq * 8;
    if (kb + KT < DIM) {                    // gfx1250 global_prefetch_b8
      __builtin_prefetch(r1 + KT, 0, 0);
      __builtin_prefetch(r3 + KT, 0, 0);
    }
    float4 f0 = ((const float4*)r1)[0], f1 = ((const float4*)r1)[1];
    b1R = make_uint4(pk_bf16x2(f0.x, f0.y), pk_bf16x2(f0.z, f0.w),
                     pk_bf16x2(f1.x, f1.y), pk_bf16x2(f1.z, f1.w));
    float4 g0 = ((const float4*)r3)[0], g1 = ((const float4*)r3)[1];
    b3R = make_uint4(pk_bf16x2(g0.x, g0.y), pk_bf16x2(g0.z, g0.w),
                     pk_bf16x2(g1.x, g1.y), pk_bf16x2(g1.z, g1.w));
  };
  auto store_B = [&](int buf) {
    *(uint4*)&B1s[buf][brow * SA + bq * 8] = b1R;
    *(uint4*)&B3s[buf][brow * SA + bq * 8] = b3R;
  };
  auto compute = [&](int buf) {
    const int c0 = (lane < 16) ? 0 : 8;     // K sub-chunk split per ISA layout
    const int lr = lane & 15;
    Frag16 a[2], b[2];
#pragma unroll
    for (int i = 0; i < 2; ++i) {
      const unsigned short* p = &As[buf][(wm * 32 + i * 16 + lr) * SA];
      a[i].q[0] = *(const uint4*)(p + c0);
      a[i].q[1] = *(const uint4*)(p + 16 + c0);
    }
#pragma unroll
    for (int j = 0; j < 2; ++j) {           // W1 pass
      const unsigned short* p = &B1s[buf][(wn * 32 + j * 16 + lr) * SA];
      b[j].q[0] = *(const uint4*)(p + c0);
      b[j].q[1] = *(const uint4*)(p + 16 + c0);
    }
#pragma unroll
    for (int i = 0; i < 2; ++i)
#pragma unroll
      for (int j = 0; j < 2; ++j)
        acc1[i][j] = __builtin_amdgcn_wmma_f32_16x16x32_bf16(
            false, a[i].v, false, b[j].v, (short)0, acc1[i][j], false, false);
#pragma unroll
    for (int j = 0; j < 2; ++j) {           // W3 pass (reuse b frags)
      const unsigned short* p = &B3s[buf][(wn * 32 + j * 16 + lr) * SA];
      b[j].q[0] = *(const uint4*)(p + c0);
      b[j].q[1] = *(const uint4*)(p + 16 + c0);
    }
#pragma unroll
    for (int i = 0; i < 2; ++i)
#pragma unroll
      for (int j = 0; j < 2; ++j)
        acc3[i][j] = __builtin_amdgcn_wmma_f32_16x16x32_bf16(
            false, a[i].v, false, b[j].v, (short)0, acc3[i][j], false, false);
  };

  const int nk = DIM / KT;
  stage_A(0, 0); load_B(0); store_B(0);
  ASYNC_WAIT();
  __syncthreads();
  int cur = 0;
  for (int kt = 0; kt < nk; ++kt) {
    if (kt + 1 < nk) { stage_A(kt + 1, cur ^ 1); load_B(kt + 1); }
    compute(cur);
    if (kt + 1 < nk) { store_B(cur ^ 1); ASYNC_WAIT(); }
    __syncthreads();
    cur ^= 1;
  }

  // epilogue: h = silu(c1) * c3, bf16 store
#pragma unroll
  for (int i = 0; i < 2; ++i)
#pragma unroll
    for (int j = 0; j < 2; ++j)
#pragma unroll
      for (int v = 0; v < 8; ++v) {
        const float c1 = acc1[i][j][v];
        const float c3 = acc3[i][j][v];
        const float h = (c1 / (1.f + __expf(-c1))) * c3;
        const int t   = rowBase + wm * 32 + i * 16 + v + ((lane >> 4) << 3);
        const int col = colBase + wn * 32 + j * 16 + (lane & 15);
        H[(size_t)t * INTER + col] = f2bf(h);
      }
}

// ------------------------------------------------- stage 2: Y (+)= (H W2^T)*combine
__global__ __launch_bounds__(256)
void moe_stage2(const unsigned short* __restrict__ Hb,   // [TOK, INTER] bf16
                const float* __restrict__ W2,            // [DIM, INTER]
                float* __restrict__ Out,                 // [TOK, DIM]
                const float* __restrict__ combine,       // [TOK, NEXP]
                int expert, int useCombine, int accum) {
  __shared__ __align__(16) unsigned short As[2][BM * SA];
  __shared__ __align__(16) unsigned short Bs[2][BN * SA];

  const int tid  = threadIdx.x;
  const int lane = tid & 31;
  const int wv   = tid >> 5;
  const int wm   = wv >> 1;
  const int wn   = wv & 1;
  const int rowBase = blockIdx.x * BM;
  const int colBase = blockIdx.y * BN;
  const int brow = tid >> 2;
  const int bq   = tid & 3;

  uint4 bR;
  v8f acc[2][2];
  v8f zero = {0.f, 0.f, 0.f, 0.f, 0.f, 0.f, 0.f, 0.f};
#pragma unroll
  for (int i = 0; i < 2; ++i)
#pragma unroll
    for (int j = 0; j < 2; ++j) acc[i][j] = zero;

  auto stage_A = [&](int kt, int buf) {
    const int kb = kt * KT;
#ifdef HAVE_ASYNC_LDS
#pragma unroll
    for (int k = 0; k < 2; ++k) {
      const int idx = tid + 256 * k;
      const int r = idx >> 2, c = idx & 3;
      async_copy16(Hb + (size_t)(rowBase + r) * INTER + kb + c * 8,
                   &As[buf][r * SA + c * 8]);
    }
#else
    const int r = tid >> 1, hc = (tid & 1) * 16;
    const uint4* ga = (const uint4*)(Hb + (size_t)(rowBase + r) * INTER + kb + hc);
    uint4* da = (uint4*)&As[buf][r * SA + hc];
    da[0] = ga[0]; da[1] = ga[1];
#endif
  };
  auto load_B = [&](int kt) {
    const int kb = kt * KT;
    const float* r2 = W2 + (size_t)(colBase + brow) * INTER + kb + bq * 8;
    if (kb + KT < INTER) __builtin_prefetch(r2 + KT, 0, 0);
    float4 f0 = ((const float4*)r2)[0], f1 = ((const float4*)r2)[1];
    bR = make_uint4(pk_bf16x2(f0.x, f0.y), pk_bf16x2(f0.z, f0.w),
                    pk_bf16x2(f1.x, f1.y), pk_bf16x2(f1.z, f1.w));
  };
  auto store_B = [&](int buf) {
    *(uint4*)&Bs[buf][brow * SA + bq * 8] = bR;
  };
  auto compute = [&](int buf) {
    const int c0 = (lane < 16) ? 0 : 8;
    const int lr = lane & 15;
    Frag16 a[2], b[2];
#pragma unroll
    for (int i = 0; i < 2; ++i) {
      const unsigned short* p = &As[buf][(wm * 32 + i * 16 + lr) * SA];
      a[i].q[0] = *(const uint4*)(p + c0);
      a[i].q[1] = *(const uint4*)(p + 16 + c0);
    }
#pragma unroll
    for (int j = 0; j < 2; ++j) {
      const unsigned short* p = &Bs[buf][(wn * 32 + j * 16 + lr) * SA];
      b[j].q[0] = *(const uint4*)(p + c0);
      b[j].q[1] = *(const uint4*)(p + 16 + c0);
    }
#pragma unroll
    for (int i = 0; i < 2; ++i)
#pragma unroll
      for (int j = 0; j < 2; ++j)
        acc[i][j] = __builtin_amdgcn_wmma_f32_16x16x32_bf16(
            false, a[i].v, false, b[j].v, (short)0, acc[i][j], false, false);
  };

  const int nk = INTER / KT;
  stage_A(0, 0); load_B(0); store_B(0);
  ASYNC_WAIT();
  __syncthreads();
  int cur = 0;
  for (int kt = 0; kt < nk; ++kt) {
    if (kt + 1 < nk) { stage_A(kt + 1, cur ^ 1); load_B(kt + 1); }
    compute(cur);
    if (kt + 1 < nk) { store_B(cur ^ 1); ASYNC_WAIT(); }
    __syncthreads();
    cur ^= 1;
  }

#pragma unroll
  for (int i = 0; i < 2; ++i)
#pragma unroll
    for (int j = 0; j < 2; ++j)
#pragma unroll
      for (int v = 0; v < 8; ++v) {
        const int t   = rowBase + wm * 32 + i * 16 + v + ((lane >> 4) << 3);
        const int col = colBase + wn * 32 + j * 16 + (lane & 15);
        const float s = useCombine ? combine[t * NEXP + expert] : 1.0f;
        const float val = acc[i][j][v] * s;
        float* o = Out + (size_t)t * DIM + col;
        if (accum) *o += val; else *o = val;
      }
}

// ---------------------------------------------------------------- launch
extern "C" void kernel_launch(void* const* d_in, const int* in_sizes, int n_in,
                              void* d_out, int out_size, void* d_ws, size_t ws_size,
                              hipStream_t stream) {
  const float* x   = (const float*)d_in[0];
  const float* gw  = (const float*)d_in[1];
  const float* gb  = (const float*)d_in[2];
  const float* w1  = (const float*)d_in[3];
  const float* w2  = (const float*)d_in[4];
  const float* w3  = (const float*)d_in[5];
  const float* sw1 = (const float*)d_in[6];
  const float* sw2 = (const float*)d_in[7];
  const float* sw3 = (const float*)d_in[8];
  float* out = (float*)d_out;

  char* ws = (char*)d_ws;
  unsigned short* Xb = (unsigned short*)ws;                          // 16 MiB
  unsigned short* Hb = (unsigned short*)(ws + ((size_t)16 << 20));   // 32 MiB
  float* combine     = (float*)(ws + ((size_t)48 << 20));            // 256 KiB

  moe_cast_bf16<<<(TOK * DIM / 4 + 255) / 256, 256, 0, stream>>>(x, Xb, TOK * DIM / 4);
  moe_gate<<<TOK / 8, 256, 0, stream>>>(x, gw, gb, combine);

  dim3 g1(TOK / BM, INTER / BN);
  dim3 g2(TOK / BM, DIM / BN);

  // shared expert first: initializes d_out (store, scale 1)
  moe_stage1<<<g1, 256, 0, stream>>>(Xb, sw1, sw3, Hb);
  moe_stage2<<<g2, 256, 0, stream>>>(Hb, sw2, out, combine, 0, 0, 0);

  for (int e = 0; e < NEXP; ++e) {
    moe_stage1<<<g1, 256, 0, stream>>>(Xb, w1 + (size_t)e * INTER * DIM,
                                       w3 + (size_t)e * INTER * DIM, Hb);
    moe_stage2<<<g2, 256, 0, stream>>>(Hb, w2 + (size_t)e * DIM * INTER,
                                       out, combine, e, 1, 1);
  }
}
